// LaCTLayer_9088150798969
// MI455X (gfx1250) — compile-verified, matrix operand order checked by reading, not run on hardware
//
#include <hip/hip_runtime.h>
#include <hip/hip_bf16.h>
#include <math.h>
#include <stdint.h>

// ---------------------------------------------------------------------------
// CDNA5 (gfx1250) wave32 WMMA helpers.  D = A(16x32 bf16) x B(32x16 bf16) + C(16x16 f32)
// A-frag: lane = M row (l&15), elements = K per ISA table (16-bit A 16x32).
// B-frag: lane = N col, elements = K (same K element map).
// C/D:    lane = N col (l&15), VGPR r = row (r + 8*(l>>4)).
// ---------------------------------------------------------------------------
typedef __bf16 bf16;
typedef __attribute__((ext_vector_type(16))) __bf16 bf16x16;
typedef __attribute__((ext_vector_type(8)))  float  f32x8;

#define TT 2048
#define HH 16
#define DH 64
#define DM 256

__device__ __forceinline__ int kmap(int i, int h) {
  int v = i >> 1;
  return ((v >> 2) << 4) + (h << 3) + ((v & 3) << 1) + (i & 1);
}

// lane indexes row, elements index K along the row (A from row-major, or B from [N][K] storage)
__device__ __forceinline__ bf16x16 frag_row(const bf16* p, int ld) {
  int l = threadIdx.x & 31, r = l & 15, h = l >> 4;
  bf16x16 f;
#pragma unroll
  for (int i = 0; i < 16; i++) f[i] = p[(size_t)r * ld + kmap(i, h)];
  return f;
}

// elements index K (row of memory), lane indexes column (B from [K][N] storage, or A^T)
__device__ __forceinline__ bf16x16 frag_col(const bf16* p, int ld) {
  int l = threadIdx.x & 31, n = l & 15, h = l >> 4;
  bf16x16 f;
#pragma unroll
  for (int i = 0; i < 16; i++) f[i] = p[(size_t)kmap(i, h) * ld + n];
  return f;
}

__device__ __forceinline__ f32x8 wmma_bf(bf16x16 a, bf16x16 b, f32x8 c) {
  return __builtin_amdgcn_wmma_f32_16x16x32_bf16(false, a, false, b, (short)0, c, false, false);
}

__device__ __forceinline__ f32x8 zero8() {
  f32x8 z = {0.f, 0.f, 0.f, 0.f, 0.f, 0.f, 0.f, 0.f};
  return z;
}

__device__ __forceinline__ float sigmf(float x) { return 1.f / (1.f + __expf(-x)); }

// gfx1250 async copy global->LDS (16B per lane per op, tracked by ASYNCcnt)
__device__ __forceinline__ void async_g2l_b128(uint32_t lds_off, uint64_t gaddr) {
  asm volatile("global_load_async_to_lds_b128 %0, %1, off"
               :: "v"(lds_off), "v"(gaddr) : "memory");
}
__device__ __forceinline__ void async_wait0() {
  asm volatile("s_wait_asynccnt 0x0" ::: "memory");
}

// ---------------------------------------------------------------------------
// Generic GEMM: Y[M][N] = X[M][K] @ W[N][K]^T + bias[N], f32 in/out, bf16 WMMA core.
// BM=BN=64, BK=32, 128 threads = 4 waves in 2x2, each wave 32x32 (2x2 WMMA tiles).
// ---------------------------------------------------------------------------
__global__ __launch_bounds__(128) void gemm_xwt_kernel(
    const float* __restrict__ X, const float* __restrict__ W,
    const float* __restrict__ bias, float* __restrict__ Y,
    int M, int N, int K)
{
  __shared__ bf16 Xs[64][40];
  __shared__ bf16 Ws[64][40];
  const int tid = threadIdx.x;
  const int wv = tid >> 5, ln = tid & 15, lh = (tid >> 4) & 1;
  const int wm = wv >> 1, wn = wv & 1;
  const int m0 = blockIdx.y * 64, n0 = blockIdx.x * 64;

  f32x8 c[2][2];
#pragma unroll
  for (int i = 0; i < 2; i++)
#pragma unroll
    for (int j = 0; j < 2; j++) c[i][j] = zero8();

  for (int k0 = 0; k0 < K; k0 += 32) {
    // stage 64x32 X and W tiles as bf16 (float4 loads)
    int r = tid >> 3;            // 0..15
    int c4 = (tid & 7) * 4;      // 0..28
#pragma unroll
    for (int rep = 0; rep < 4; rep++) {
      int row = rep * 16 + r;
      float4 xv = *reinterpret_cast<const float4*>(X + (size_t)(m0 + row) * K + k0 + c4);
      Xs[row][c4 + 0] = (bf16)xv.x; Xs[row][c4 + 1] = (bf16)xv.y;
      Xs[row][c4 + 2] = (bf16)xv.z; Xs[row][c4 + 3] = (bf16)xv.w;
      float4 wvv = make_float4(0.f, 0.f, 0.f, 0.f);
      if (n0 + row < N)
        wvv = *reinterpret_cast<const float4*>(W + (size_t)(n0 + row) * K + k0 + c4);
      Ws[row][c4 + 0] = (bf16)wvv.x; Ws[row][c4 + 1] = (bf16)wvv.y;
      Ws[row][c4 + 2] = (bf16)wvv.z; Ws[row][c4 + 3] = (bf16)wvv.w;
    }
    if (k0 + 32 < K) { // hint next tile into cache (global_prefetch on gfx1250)
      __builtin_prefetch(X + (size_t)(m0 + r) * K + k0 + 32, 0, 1);
      __builtin_prefetch(W + (size_t)(n0 + r) * K + k0 + 32, 0, 1);
    }
    __syncthreads();

    bf16x16 a0 = frag_row(&Xs[wm * 32 +  0][0], 40);
    bf16x16 a1 = frag_row(&Xs[wm * 32 + 16][0], 40);
    bf16x16 b0 = frag_row(&Ws[wn * 32 +  0][0], 40);
    bf16x16 b1 = frag_row(&Ws[wn * 32 + 16][0], 40);
    c[0][0] = wmma_bf(a0, b0, c[0][0]);
    c[0][1] = wmma_bf(a0, b1, c[0][1]);
    c[1][0] = wmma_bf(a1, b0, c[1][0]);
    c[1][1] = wmma_bf(a1, b1, c[1][1]);
    __syncthreads();
  }

#pragma unroll
  for (int i = 0; i < 2; i++)
#pragma unroll
    for (int j = 0; j < 2; j++) {
      int col = n0 + wn * 32 + j * 16 + ln;
      if (col < N) {
        float bv = bias[col];
#pragma unroll
        for (int r = 0; r < 8; r++) {
          int row = m0 + wm * 32 + i * 16 + r + 8 * lh;
          Y[(size_t)row * N + col] = c[i][j][r] + bv;
        }
      }
    }
}

// ---------------------------------------------------------------------------
// Prep: RoPE, silu, L2-norm + scale/bias, softplus(lr). One thread per (h,t).
// ---------------------------------------------------------------------------
__global__ __launch_bounds__(256) void prep_kernel(
    const float* __restrict__ qkv, const float* __restrict__ lrraw,
    const float* __restrict__ cosb, const float* __restrict__ sinb,
    const float* __restrict__ qsc, const float* __restrict__ qbi,
    const float* __restrict__ ksc, const float* __restrict__ kbi,
    bf16* __restrict__ qrb, bf16* __restrict__ krb, bf16* __restrict__ vb,
    bf16* __restrict__ qnb, bf16* __restrict__ knb, bf16* __restrict__ vsb,
    float* __restrict__ lrc)
{
  int idx = blockIdx.x * blockDim.x + threadIdx.x;
  if (idx >= HH * TT) return;
  int h = idx & (HH - 1);
  int t = idx >> 4;
  const float* base = qkv + (size_t)t * (3 * HH * DH) + h * DH;
  const float* cr = cosb + (size_t)t * DH;
  const float* sr = sinb + (size_t)t * DH;
  size_t ho = ((size_t)h * TT + t) * DH;
  float a[DH];

  // ---- Q ----
  for (int i = 0; i < DH; i++) a[i] = base[i];
  for (int i = 0; i < 32; i++) qrb[ho + i]      = (bf16)(a[i] * cr[i] - a[i + 32] * sr[i]);
  for (int i = 32; i < DH; i++) qrb[ho + i]     = (bf16)(a[i] * cr[i] + a[i - 32] * sr[i]);
  float ss = 0.f;
  for (int i = 0; i < DH; i++) { float s = a[i] * sigmf(a[i]); a[i] = s; ss += s * s; }
  float inv = 1.f / (sqrtf(ss) + 1e-6f);
  for (int i = 0; i < DH; i++)
    qnb[ho + i] = (bf16)(a[i] * inv * qsc[h * DH + i] + qbi[h * DH + i]);

  // ---- K ----
  for (int i = 0; i < DH; i++) a[i] = base[HH * DH + i];
  for (int i = 0; i < 32; i++) krb[ho + i]      = (bf16)(a[i] * cr[i] - a[i + 32] * sr[i]);
  for (int i = 32; i < DH; i++) krb[ho + i]     = (bf16)(a[i] * cr[i] + a[i - 32] * sr[i]);
  ss = 0.f;
  for (int i = 0; i < DH; i++) { float s = a[i] * sigmf(a[i]); a[i] = s; ss += s * s; }
  inv = 1.f / (sqrtf(ss) + 1e-6f);
  for (int i = 0; i < DH; i++)
    knb[ho + i] = (bf16)(a[i] * inv * ksc[h * DH + i] + kbi[h * DH + i]);

  // ---- V ----
  for (int i = 0; i < DH; i++) {
    float v = base[2 * HH * DH + i];
    vb[ho + i]  = (bf16)v;
    vsb[ho + i] = (bf16)(v * sigmf(v));
  }

  // ---- lr softplus ----
  for (int j = 0; j < 3; j++) {
    float xv = lrraw[(size_t)t * (3 * HH) + h * 3 + j];
    lrc[((size_t)h * TT + t) * 3 + j] = (xv > 20.f) ? xv : log1pf(__expf(xv));
  }
}

// ---------------------------------------------------------------------------
// Causal flash attention: one wave per (16-query tile, head).
// K/V tiles staged into LDS with gfx1250 async copies (ASYNCcnt), WMMA core.
// ---------------------------------------------------------------------------
__global__ __launch_bounds__(32) void attn_kernel(
    const bf16* __restrict__ qrb, const bf16* __restrict__ krb,
    const bf16* __restrict__ vb, float* __restrict__ attn)
{
  const int qt = blockIdx.x, hh = blockIdx.y;
  const int q0 = qt * 16;
  const int l = threadIdx.x & 31, ln = l & 15, lh = l >> 4;
  const float scale = 0.125f; // 1/sqrt(64)
  __shared__ bf16 Ks[32 * 64];
  __shared__ bf16 Vs[32 * 64];
  __shared__ bf16 pscr[16][40];
  const uint32_t ldsK = (uint32_t)(uintptr_t)&Ks[0] + (uint32_t)l * 128u;
  const uint32_t ldsV = (uint32_t)(uintptr_t)&Vs[0] + (uint32_t)l * 128u;

  const bf16* qp = qrb + ((size_t)hh * TT + q0) * DH;
  bf16x16 aq0 = frag_row(qp, DH);
  bf16x16 aq1 = frag_row(qp + 32, DH);

  f32x8 o[4];
#pragma unroll
  for (int i = 0; i < 4; i++) o[i] = zero8();
  float mrow[8], lrow[8];
#pragma unroll
  for (int r = 0; r < 8; r++) { mrow[r] = -3.0e38f; lrow[r] = 0.f; }

  const int nblk = q0 / 32 + 1;
  for (int b = 0; b < nblk; b++) {
    const int k0 = b * 32;
    // async-stage K/V tiles (32 rows x 128B): lane l copies row k0+l
    {
      uint64_t gK = (uint64_t)(uintptr_t)(krb + ((size_t)hh * TT + k0 + l) * DH);
      uint64_t gV = (uint64_t)(uintptr_t)(vb  + ((size_t)hh * TT + k0 + l) * DH);
#pragma unroll
      for (int j = 0; j < 8; j++) {
        async_g2l_b128(ldsK + (uint32_t)(j * 16), gK + (uint64_t)(j * 16));
        async_g2l_b128(ldsV + (uint32_t)(j * 16), gV + (uint64_t)(j * 16));
      }
      async_wait0();
    }
    f32x8 s0 = zero8(), s1 = zero8();
    {
      bf16x16 b00 = frag_row(Ks, DH);
      bf16x16 b01 = frag_row(Ks + 32, DH);
      s0 = wmma_bf(aq0, b00, s0); s0 = wmma_bf(aq1, b01, s0);
      bf16x16 b10 = frag_row(Ks + 16 * DH, DH);
      bf16x16 b11 = frag_row(Ks + 16 * DH + 32, DH);
      s1 = wmma_bf(aq0, b10, s1); s1 = wmma_bf(aq1, b11, s1);
    }
#pragma unroll
    for (int r = 0; r < 8; r++) {
      int q = q0 + r + 8 * lh;
      float v0 = s0[r] * scale; if (k0 + ln      > q) v0 = -3.0e38f;
      float v1 = s1[r] * scale; if (k0 + 16 + ln > q) v1 = -3.0e38f;
      float mx = fmaxf(v0, v1);
#pragma unroll
      for (int off = 1; off < 16; off <<= 1) mx = fmaxf(mx, __shfl_xor(mx, off, 32));
      float mn = fmaxf(mrow[r], mx);
      float p0 = __expf(v0 - mn);
      float p1 = __expf(v1 - mn);
      float alpha = __expf(mrow[r] - mn);
      float sp = p0 + p1;
#pragma unroll
      for (int off = 1; off < 16; off <<= 1) sp += __shfl_xor(sp, off, 32);
      lrow[r] = lrow[r] * alpha + sp;
      mrow[r] = mn;
#pragma unroll
      for (int dt = 0; dt < 4; dt++) o[dt][r] *= alpha;
      pscr[r + 8 * lh][ln]      = (bf16)p0;
      pscr[r + 8 * lh][16 + ln] = (bf16)p1;
    }
    asm volatile("s_wait_dscnt 0x0" ::: "memory"); // wave-local LDS RAW
    bf16x16 ap = frag_row(&pscr[0][0], 40);        // K = 32 keys
#pragma unroll
    for (int dt = 0; dt < 4; dt++) {
      bf16x16 bv = frag_col(Vs + dt * 16, DH);
      o[dt] = wmma_bf(ap, bv, o[dt]);
    }
    // next iteration's async stores must not race this iteration's reads:
    asm volatile("s_wait_dscnt 0x0" ::: "memory");
  }
#pragma unroll
  for (int dt = 0; dt < 4; dt++)
#pragma unroll
    for (int r = 0; r < 8; r++) {
      int t = q0 + r + 8 * lh;
      attn[(size_t)t * (HH * DH) + hh * DH + dt * 16 + ln] = o[dt][r] / lrow[r];
    }
}

// ---------------------------------------------------------------------------
// TTT chunked scan: one 256-thread block per head, 8 sequential chunks.
// bf16 shadow weights in LDS; f32 master weights + momentum in global scratch.
// ---------------------------------------------------------------------------
__global__ __launch_bounds__(256) void ttt_kernel(
    const float* __restrict__ w1_in, const float* __restrict__ w2_in, const float* __restrict__ w3_in,
    const bf16* __restrict__ qnb, const bf16* __restrict__ knb, const bf16* __restrict__ vsb,
    const float* __restrict__ lrc,
    float* __restrict__ w1f, float* __restrict__ w2f, float* __restrict__ w3f,
    float* __restrict__ m1f, float* __restrict__ m2f, float* __restrict__ m3f,
    float* __restrict__ tttb)
{
  const int hh = blockIdx.x;
  const int tid = threadIdx.x;
  const int wv = tid >> 5, ln = tid & 15, lh = (tid >> 4) & 1;
  const int mw0 = wv * 32;

  __shared__ bf16 w1s[DH * DM];   // [d][m]
  __shared__ bf16 w2s[DM * DH];   // [m][d]
  __shared__ bf16 w3s[DH * DM];   // [d][m]
  __shared__ bf16 scr[32 * DM];   // staging [c][m]
  __shared__ bf16 scr2[32 * DH];  // dout*lr2 [c][d]
  __shared__ float lrs[256 * 3];

  const size_t HW = (size_t)hh * (DH * DM);
  for (int j = tid; j < DH * DM; j += 256) {
    w1s[j] = (bf16)w1_in[HW + j];
    w3s[j] = (bf16)w3_in[HW + j];
    w2s[j] = (bf16)w2_in[HW + j];
  }
  __syncthreads();

  const bf16* kh = knb + (size_t)hh * TT * DH;
  const bf16* vh = vsb + (size_t)hh * TT * DH;
  const bf16* qh = qnb + (size_t)hh * TT * DH;
  const float* lrh = lrc + (size_t)hh * TT * 3;

  for (int cc = 0; cc < 8; cc++) {
    const int t0 = cc * 256;
    for (int j = tid; j < 256 * 3; j += 256) lrs[j] = lrh[(size_t)t0 * 3 + j];
    __syncthreads();

    f32x8 dW1[8], dW3[8], dW2a[8];
#pragma unroll
    for (int i = 0; i < 8; i++) { dW1[i] = zero8(); dW3[i] = zero8(); dW2a[i] = zero8(); }

    for (int ct = 0; ct < 8; ct++) {
      const int c0 = t0 + ct * 32;
      bf16x16 aK[2][2], aV[2][2];
#pragma unroll
      for (int mb = 0; mb < 2; mb++)
#pragma unroll
        for (int ks = 0; ks < 2; ks++) {
          aK[mb][ks] = frag_row(kh + (size_t)(c0 + mb * 16) * DH + ks * 32, DH);
          aV[mb][ks] = frag_row(vh + (size_t)(c0 + mb * 16) * DH + ks * 32, DH);
        }
      f32x8 G[2][2], H3[2][2], DHt[2][2];
#pragma unroll
      for (int mb = 0; mb < 2; mb++)
#pragma unroll
        for (int nt = 0; nt < 2; nt++) { G[mb][nt] = zero8(); H3[mb][nt] = zero8(); DHt[mb][nt] = zero8(); }

#pragma unroll
      for (int ks = 0; ks < 2; ks++)
#pragma unroll
        for (int nt = 0; nt < 2; nt++) {
          bf16x16 b1 = frag_col(w1s + (size_t)ks * 32 * DM + mw0 + nt * 16, DM);
          bf16x16 b3 = frag_col(w3s + (size_t)ks * 32 * DM + mw0 + nt * 16, DM);
          bf16x16 b2 = frag_row(w2s + (size_t)(mw0 + nt * 16) * DH + ks * 32, DH);
#pragma unroll
          for (int mb = 0; mb < 2; mb++) {
            G[mb][nt]   = wmma_bf(aK[mb][ks], b1, G[mb][nt]);
            H3[mb][nt]  = wmma_bf(aK[mb][ks], b3, H3[mb][nt]);
            DHt[mb][nt] = wmma_bf(aV[mb][ks], b2, DHt[mb][nt]);
          }
        }

      // elementwise: dg*lr1 -> G, dh3*lr3 -> H3, hid -> DHt
#pragma unroll
      for (int mb = 0; mb < 2; mb++)
#pragma unroll
        for (int nt = 0; nt < 2; nt++)
#pragma unroll
          for (int r = 0; r < 8; r++) {
            int cl = ct * 32 + mb * 16 + r + 8 * lh;
            float lr1 = lrs[cl * 3 + 0], lr3v = lrs[cl * 3 + 2];
            float g = G[mb][nt][r], h3v = H3[mb][nt][r];
            float dh = -DHt[mb][nt][r];
            float sg = sigmf(g);
            float silu = g * sg;
            float hid = silu * h3v;
            float dh3 = dh * silu;
            float dgv = dh * h3v * (sg * (1.f + g * (1.f - sg)));
            G[mb][nt][r]   = dgv * lr1;
            H3[mb][nt][r]  = dh3 * lr3v;
            DHt[mb][nt][r] = hid;
          }
      // scr2 = dout*lr2 = -vs*lr2
      for (int j = tid; j < 32 * DH; j += 256) {
        int c = j >> 6, d = j & 63;
        float lr2 = lrs[(ct * 32 + c) * 3 + 1];
        scr2[j] = (bf16)(-(float)vh[(size_t)(c0 + c) * DH + d] * lr2);
      }
      // phase 1: dg*lr1 -> scr; dW1 += K^T @ scr
#pragma unroll
      for (int mb = 0; mb < 2; mb++)
#pragma unroll
        for (int nt = 0; nt < 2; nt++)
#pragma unroll
          for (int r = 0; r < 8; r++)
            scr[(size_t)(mb * 16 + r + 8 * lh) * DM + mw0 + nt * 16 + ln] = (bf16)G[mb][nt][r];
      __syncthreads();
      bf16x16 aKT[4];
#pragma unroll
      for (int db = 0; db < 4; db++) aKT[db] = frag_col(kh + (size_t)c0 * DH + db * 16, DH);
#pragma unroll
      for (int nt = 0; nt < 2; nt++) {
        bf16x16 bD = frag_col(scr + mw0 + nt * 16, DM);
#pragma unroll
        for (int db = 0; db < 4; db++)
          dW1[db * 2 + nt] = wmma_bf(aKT[db], bD, dW1[db * 2 + nt]);
      }
      __syncthreads();
      // phase 2: dh3*lr3 -> scr; dW3 += K^T @ scr
#pragma unroll
      for (int mb = 0; mb < 2; mb++)
#pragma unroll
        for (int nt = 0; nt < 2; nt++)
#pragma unroll
          for (int r = 0; r < 8; r++)
            scr[(size_t)(mb * 16 + r + 8 * lh) * DM + mw0 + nt * 16 + ln] = (bf16)H3[mb][nt][r];
      __syncthreads();
#pragma unroll
      for (int nt = 0; nt < 2; nt++) {
        bf16x16 bD = frag_col(scr + mw0 + nt * 16, DM);
#pragma unroll
        for (int db = 0; db < 4; db++)
          dW3[db * 2 + nt] = wmma_bf(aKT[db], bD, dW3[db * 2 + nt]);
      }
      __syncthreads();
      // phase 3: hid -> scr; dW2 += HID^T @ (dout*lr2)
#pragma unroll
      for (int mb = 0; mb < 2; mb++)
#pragma unroll
        for (int nt = 0; nt < 2; nt++)
#pragma unroll
          for (int r = 0; r < 8; r++)
            scr[(size_t)(mb * 16 + r + 8 * lh) * DM + mw0 + nt * 16 + ln] = (bf16)DHt[mb][nt][r];
      __syncthreads();
#pragma unroll
      for (int mb = 0; mb < 2; mb++) {
        bf16x16 aH = frag_col(scr + mw0 + mb * 16, DM);
#pragma unroll
        for (int dt = 0; dt < 4; dt++) {
          bf16x16 bO = frag_col(scr2 + dt * 16, DH);
          dW2a[mb * 4 + dt] = wmma_bf(aH, bO, dW2a[mb * 4 + dt]);
        }
      }
      __syncthreads();
    } // ct

    // momentum + weight update (each wave owns its DM slice)
#pragma unroll
    for (int db = 0; db < 4; db++)
#pragma unroll
      for (int nt = 0; nt < 2; nt++)
#pragma unroll
        for (int r = 0; r < 8; r++) {
          int d = db * 16 + r + 8 * lh;
          int m = mw0 + nt * 16 + ln;
          size_t idx = HW + (size_t)d * DM + m;
          float dv = dW1[db * 2 + nt][r];
          float mo = (cc == 0) ? 0.f : m1f[idx];
          float mn = 0.9f * mo + dv; m1f[idx] = mn;
          float wc = (cc == 0) ? w1_in[idx] : w1f[idx];
          float wn = wc - mn; w1f[idx] = wn;
          w1s[(size_t)d * DM + m] = (bf16)wn;

          dv = dW3[db * 2 + nt][r];
          mo = (cc == 0) ? 0.f : m3f[idx];
          mn = 0.9f * mo + dv; m3f[idx] = mn;
          wc = (cc == 0) ? w3_in[idx] : w3f[idx];
          wn = wc - mn; w3f[idx] = wn;
          w3s[(size_t)d * DM + m] = (bf16)wn;
        }
#pragma unroll
    for (int mb = 0; mb < 2; mb++)
#pragma unroll
      for (int dt = 0; dt < 4; dt++)
#pragma unroll
        for (int r = 0; r < 8; r++) {
          int m = mw0 + mb * 16 + r + 8 * lh;
          int d = dt * 16 + ln;
          size_t idx = HW + (size_t)m * DH + d;
          float dv = dW2a[mb * 4 + dt][r];
          float mo = (cc == 0) ? 0.f : m2f[idx];
          float mn = 0.9f * mo + dv; m2f[idx] = mn;
          float wc = (cc == 0) ? w2_in[idx] : w2f[idx];
          float wn = wc - mn; w2f[idx] = wn;
          w2s[(size_t)m * DH + d] = (bf16)wn;
        }
    __syncthreads();

    // Q pass with updated weights
    for (int ct = 0; ct < 8; ct++) {
      const int c0 = t0 + ct * 32;
      bf16x16 aQ[2][2];
#pragma unroll
      for (int mb = 0; mb < 2; mb++)
#pragma unroll
        for (int ks = 0; ks < 2; ks++)
          aQ[mb][ks] = frag_row(qh + (size_t)(c0 + mb * 16) * DH + ks * 32, DH);
      f32x8 Gq[2][2], Hq[2][2];
#pragma unroll
      for (int mb = 0; mb < 2; mb++)
#pragma unroll
        for (int nt = 0; nt < 2; nt++) { Gq[mb][nt] = zero8(); Hq[mb][nt] = zero8(); }
#pragma unroll
      for (int ks = 0; ks < 2; ks++)
#pragma unroll
        for (int nt = 0; nt < 2; nt++) {
          bf16x16 b1 = frag_col(w1s + (size_t)ks * 32 * DM + mw0 + nt * 16, DM);
          bf16x16 b3 = frag_col(w3s + (size_t)ks * 32 * DM + mw0 + nt * 16, DM);
#pragma unroll
          for (int mb = 0; mb < 2; mb++) {
            Gq[mb][nt] = wmma_bf(aQ[mb][ks], b1, Gq[mb][nt]);
            Hq[mb][nt] = wmma_bf(aQ[mb][ks], b3, Hq[mb][nt]);
          }
        }
      __syncthreads();
#pragma unroll
      for (int mb = 0; mb < 2; mb++)
#pragma unroll
        for (int nt = 0; nt < 2; nt++)
#pragma unroll
          for (int r = 0; r < 8; r++) {
            float g = Gq[mb][nt][r];
            scr[(size_t)(mb * 16 + r + 8 * lh) * DM + mw0 + nt * 16 + ln] =
                (bf16)(g * sigmf(g) * Hq[mb][nt][r]);
          }
      __syncthreads();
      // O = hid_q @ W2 : wave computes (cb, dt2) 16x16 output tile, K=256
      const int cb = wv >> 2, dt2 = wv & 3;
      f32x8 o = zero8();
#pragma unroll
      for (int ks = 0; ks < 8; ks++) {
        bf16x16 aP = frag_row(scr + (size_t)(cb * 16) * DM + ks * 32, DM);
        bf16x16 bW = frag_col(w2s + (size_t)ks * 32 * DH + dt2 * 16, DH);
        o = wmma_bf(aP, bW, o);
      }
#pragma unroll
      for (int r = 0; r < 8; r++) {
        int t = c0 + cb * 16 + r + 8 * lh;
        tttb[(size_t)t * (HH * DH) + hh * DH + dt2 * 16 + ln] = o[r];
      }
    }
    __syncthreads();
  } // cc
}

// ---------------------------------------------------------------------------
// Epilogue 1: RMS-norm over 1024, per-head 64x64 gate matmul, silu gate, + attn
// ---------------------------------------------------------------------------
__global__ __launch_bounds__(256) void epi1_kernel(
    const float* __restrict__ tttb, const float* __restrict__ attn,
    const float* __restrict__ rms_w, const float* __restrict__ sph_w,
    const float* __restrict__ sph_b, float* __restrict__ pre)
{
  const int t = blockIdx.x, tid = threadIdx.x;
  __shared__ float y[1024];
  __shared__ float partial[8];
  float ss = 0.f;
  for (int j = tid; j < 1024; j += 256) {
    float v = tttb[(size_t)t * 1024 + j];
    y[j] = v; ss += v * v;
  }
#pragma unroll
  for (int off = 1; off < 32; off <<= 1) ss += __shfl_xor(ss, off, 32);
  if ((tid & 31) == 0) partial[tid >> 5] = ss;
  __syncthreads();
  float tot = 0.f;
#pragma unroll
  for (int i = 0; i < 8; i++) tot += partial[i];
  float sc = rsqrtf(tot * (1.f / 1024.f) + 1e-5f);
  for (int j = tid; j < 1024; j += 256) y[j] = y[j] * sc * rms_w[j];
  __syncthreads();
#pragma unroll
  for (int p = 0; p < 4; p++) {
    int oi = tid + p * 256;
    int h = oi >> 6, d = oi & 63;
    float s = sph_b[d];
    const float* wrow = sph_w + (size_t)d * 64;
    const float* yh = y + h * 64;
    for (int e = 0; e < 64; e++) s += yh[e] * wrow[e];
    float sil = s * sigmf(s);
    pre[(size_t)t * 1024 + oi] = attn[(size_t)t * 1024 + oi] + y[oi] * sil;
  }
}

// ---------------------------------------------------------------------------
// Host driver
// ---------------------------------------------------------------------------
extern "C" void kernel_launch(void* const* d_in, const int* in_sizes, int n_in,
                              void* d_out, int out_size, void* d_ws, size_t ws_size,
                              hipStream_t stream) {
  (void)in_sizes; (void)n_in; (void)out_size; (void)ws_size;
  const float* x       = (const float*)d_in[0];
  // d_in[1] attn_mask: causal mask reproduced in-kernel
  const float* cosb    = (const float*)d_in[2];
  const float* sinb    = (const float*)d_in[3];
  const float* qkv_w   = (const float*)d_in[4];
  const float* qkv_b   = (const float*)d_in[5];
  const float* o_w     = (const float*)d_in[6];
  const float* o_b     = (const float*)d_in[7];
  const float* rms_w   = (const float*)d_in[8];
  const float* q_scale = (const float*)d_in[9];
  const float* q_bias  = (const float*)d_in[10];
  const float* k_scale = (const float*)d_in[11];
  const float* k_bias  = (const float*)d_in[12];
  const float* sph_w   = (const float*)d_in[13];
  const float* sph_b   = (const float*)d_in[14];
  const float* w1      = (const float*)d_in[15];
  const float* w2      = (const float*)d_in[16];
  const float* w3      = (const float*)d_in[17];
  const float* lr_w    = (const float*)d_in[18];
  const float* lr_b    = (const float*)d_in[19];

  char* ws = (char*)d_ws;
  size_t off = 0;
  auto take = [&](size_t bytes) { size_t p = off; off += (bytes + 255) & ~(size_t)255; return p; };

  float* qkvbuf = (float*)(ws + take((size_t)TT * 3072 * 4));
  float* lrraw  = (float*)(ws + take((size_t)TT * 48 * 4));
  bf16*  qrb    = (bf16*) (ws + take((size_t)HH * TT * DH * 2));
  bf16*  krb    = (bf16*) (ws + take((size_t)HH * TT * DH * 2));
  bf16*  vbuf   = (bf16*) (ws + take((size_t)HH * TT * DH * 2));
  bf16*  qnb    = (bf16*) (ws + take((size_t)HH * TT * DH * 2));
  bf16*  knb    = (bf16*) (ws + take((size_t)HH * TT * DH * 2));
  bf16*  vsb    = (bf16*) (ws + take((size_t)HH * TT * DH * 2));
  float* lrc    = (float*)(ws + take((size_t)HH * TT * 3 * 4));
  float* attn   = (float*)(ws + take((size_t)TT * 1024 * 4));
  float* tttb   = (float*)(ws + take((size_t)TT * 1024 * 4));
  float* pre    = (float*)(ws + take((size_t)TT * 1024 * 4));
  float* w1f    = (float*)(ws + take((size_t)HH * DH * DM * 4));
  float* w2f    = (float*)(ws + take((size_t)HH * DH * DM * 4));
  float* w3f    = (float*)(ws + take((size_t)HH * DH * DM * 4));
  float* m1f    = (float*)(ws + take((size_t)HH * DH * DM * 4));
  float* m2f    = (float*)(ws + take((size_t)HH * DH * DM * 4));
  float* m3f    = (float*)(ws + take((size_t)HH * DH * DM * 4));

  // 1) qkv = x @ qkv_w^T + b   (2048 x 3072 x 1024)
  gemm_xwt_kernel<<<dim3(3072 / 64, TT / 64), 128, 0, stream>>>(
      x, qkv_w, qkv_b, qkvbuf, TT, 3072, 1024);
  // 2) lr_raw = x @ lr_w^T + b (2048 x 48 x 1024)
  gemm_xwt_kernel<<<dim3(1, TT / 64), 128, 0, stream>>>(
      x, lr_w, lr_b, lrraw, TT, 48, 1024);
  // 3) RoPE / silu / norm / softplus prep
  prep_kernel<<<(HH * TT) / 256, 256, 0, stream>>>(
      qkvbuf, lrraw, cosb, sinb, q_scale, q_bias, k_scale, k_bias,
      qrb, krb, vbuf, qnb, knb, vsb, lrc);
  // 4) causal flash attention (async K/V staging)
  attn_kernel<<<dim3(TT / 16, HH), 32, 0, stream>>>(qrb, krb, vbuf, attn);
  // 5) TTT chunked scan (one block per head)
  ttt_kernel<<<HH, 256, 0, stream>>>(
      w1, w2, w3, qnb, knb, vsb, lrc, w1f, w2f, w3f, m1f, m2f, m3f, tttb);
  // 6) RMS + scale_ph gate + attn add
  epi1_kernel<<<TT, 256, 0, stream>>>(tttb, attn, rms_w, sph_w, sph_b, pre);
  // 7) out = pre @ o_proj_w^T + b (2048 x 1024 x 1024)
  gemm_xwt_kernel<<<dim3(1024 / 64, TT / 64), 128, 0, stream>>>(
      pre, o_w, o_b, (float*)d_out, TT, 1024, 1024);
}
